// CodeBook_10806137717503
// MI455X (gfx1250) — compile-verified, matrix operand order checked by reading, not run on hardware
//
#include <hip/hip_runtime.h>

// ---------------------------------------------------------------------------
// VQ-VAE CodeBook eval forward for MI455X (gfx1250, wave32, WMMA)
//   z: [64,256,32,32] f32, codebook: [1024,256] f32
//   out: z_q [16777216] | indices-as-float [65536] | loss [1] | perplexity [1]
//
// Distance matmul done as split-bf16 WMMA (hi*hi + hi*lo + lo*hi) giving
// ~fp32-equivalent ordering at bf16 matrix rates. Two independent WMMA
// accumulator chains hide the D->C accumulation latency.
// ---------------------------------------------------------------------------

typedef __attribute__((ext_vector_type(16))) __bf16 bf16x16;
typedef __attribute__((ext_vector_type(8)))  float  f32x8;

#define NB      64
#define DDIM    256
#define HW      1024
#define NROWS   65536      // NB*HW
#define KCODES  1024
#define BETA    0.25f

// ws layout (dwords)
#define WS_B      0         // 262144 dwords: swizzled bf16 hi/lo B fragments
#define WS_BIAS   262144    // 1024 floats: 0.5*||e||^2
#define WS_CNT    263168    // 1024 uints: cluster counts
#define WS_LOSS   264192    // 1 float: sum of squared diffs

__device__ __forceinline__ unsigned short f2bf(float x) {
    unsigned u = __float_as_uint(x);
    unsigned r = (u + 0x7FFFu + ((u >> 16) & 1u)) >> 16;   // RNE
    return (unsigned short)r;
}
__device__ __forceinline__ float bf2f(unsigned short h) {
    return __uint_as_float(((unsigned)h) << 16);
}

// ---------------------------------------------------------------------------
// Kernel 1: convert codebook -> WMMA-B-fragment-swizzled bf16 hi/lo in ws,
// compute per-code bias, zero counts/loss. Grid: 64 blocks (one per 16-code
// block "cb") x 256 threads.
// B fragment layout (16x16x32 bf16, K=d-chunk, N=code):
//   lane = N + 16*half; reg j holds K pair; K = (j<4?0:16) + half*8 + (j&3)*2
// ---------------------------------------------------------------------------
__global__ __launch_bounds__(256) void vq_prep(
    const float* __restrict__ cbw, unsigned* __restrict__ wsB,
    float* __restrict__ wsBias, unsigned* __restrict__ counts,
    float* __restrict__ lossAcc)
{
    const int cb   = blockIdx.x;
    const int t    = threadIdx.x;
    const int lane = t & 31;
    const int dc   = t >> 5;                 // 0..7 (d chunk of 32)
    const int code = (cb << 4) + (lane & 15);
    const int half = lane >> 4;

    unsigned* dst = wsB + cb * 4096 + dc * 512 + lane * 16;
#pragma unroll
    for (int j = 0; j < 8; ++j) {
        const int kb = ((j < 4) ? 0 : 16) + half * 8 + (j & 3) * 2;
        const int d  = (dc << 5) + kb;
        float x0 = cbw[(code << 8) + d];
        float x1 = cbw[(code << 8) + d + 1];
        unsigned short h0 = f2bf(x0), h1 = f2bf(x1);
        unsigned short l0 = f2bf(x0 - bf2f(h0));
        unsigned short l1 = f2bf(x1 - bf2f(h1));
        dst[j]     = (unsigned)h0 | ((unsigned)h1 << 16);
        dst[8 + j] = (unsigned)l0 | ((unsigned)l1 << 16);
    }
    if (t < 16) {                            // bias for 16 codes of this cb
        const int c2 = (cb << 4) + t;
        float s = 0.f;
        for (int d = 0; d < DDIM; ++d) { float e = cbw[(c2 << 8) + d]; s += e * e; }
        wsBias[c2] = 0.5f * s;
    }
    if (cb == 0) {
        for (int i = t; i < KCODES; i += 256) counts[i] = 0u;
        if (t == 0) *lossAcc = 0.f;
    }
}

// ---------------------------------------------------------------------------
// Kernel 2: main. 512 blocks x 256 threads (8 waves). Each block: 128 rows.
//   Phase 1: stage z tile -> LDS as swizzled bf16 hi/lo A-fragments (128 KB)
//   Phase 2: each wave = 16 rows, scan all 64 code-blocks with split-bf16
//            WMMA on two accumulator chains, running argmax, lane reduce
//   Phase 3: gather z_q, straight-through output, loss accumulation
// ---------------------------------------------------------------------------
__global__ __launch_bounds__(256) void vq_main(
    const float* __restrict__ z, const float* __restrict__ cbw,
    const unsigned* __restrict__ wsB, const float* __restrict__ wsBias,
    unsigned* __restrict__ counts, float* __restrict__ lossAcc,
    float* __restrict__ outZq, float* __restrict__ outIdx)
{
    extern __shared__ unsigned smem[];
    unsigned* sA   = smem;                         // 32768 dwords (128 KB)
    int*      sIdx = (int*)(smem + 32768);         // 128
    float*    sRed = (float*)(smem + 32768 + 128); // 256

    const int t        = threadIdx.x;
    const int blk      = blockIdx.x;
    const int b        = blk >> 3;           // batch index
    const int hwBase   = (blk & 7) << 7;     // 128-row window inside H*W
    const int tileBase = blk << 7;           // first flat row n

    // ---- Phase 1: coalesced z load -> hi/lo bf16 -> swizzled A fragments
    {
        const int r     = t & 127;           // row in tile
        const int dHalf = t >> 7;            // 0/1
        const int zrow  = b << 8;            // b*256
        unsigned short* sh = (unsigned short*)sA;
#pragma unroll 4
        for (int i = 0; i < 128; ++i) {
            const int d = (i << 1) + dHalf;
            float x = z[((zrow + d) << 10) + hwBase + r];
            unsigned short h = f2bf(x);
            unsigned short l = f2bf(x - bf2f(h));
            // A-fragment swizzle (16x16x32 bf16): lane = M + 16*half,
            // reg j holds K pair; K = (j<4?0:16) + half*8 + (j&3)*2 + s
            const int strip = r >> 4, M = r & 15, dc = d >> 5, k = d & 31;
            const int half = (k >> 3) & 1;
            const int j    = ((k & 16) >> 2) + ((k & 7) >> 1);
            const int lane = M + (half << 4);
            const int e    = (((strip * 8 + dc) * 32 + lane) * 16 + j) * 2 + (k & 1);
            sh[e]      = h;                  // hi dwords 0..7
            sh[e + 16] = l;                  // lo dwords 8..15
        }
    }
    __syncthreads();

    // ---- Phase 2: WMMA argmax scan
    const int lane  = t & 31;
    const int strip = t >> 5;                // wave id = 16-row strip

    bf16x16 Ahi[8], Alo[8];
#pragma unroll
    for (int dc = 0; dc < 8; ++dc) {
        const unsigned* p = sA + ((strip * 8 + dc) * 32 + lane) * 16;
        Ahi[dc] = *(const bf16x16*)p;
        Alo[dc] = *(const bf16x16*)(p + 8);
    }

    f32x8 bs = { -3.0e38f, -3.0e38f, -3.0e38f, -3.0e38f,
                 -3.0e38f, -3.0e38f, -3.0e38f, -3.0e38f };
    int bc[8] = { 0, 0, 0, 0, 0, 0, 0, 0 };

#pragma unroll 1
    for (int cb = 0; cb < 64; ++cb) {
        // Two independent accumulation chains (even dc -> acc0, odd -> acc1)
        // so dependent-WMMA latency of one chain is filled by the other.
        f32x8 acc0 = { 0.f, 0.f, 0.f, 0.f, 0.f, 0.f, 0.f, 0.f };
        f32x8 acc1 = { 0.f, 0.f, 0.f, 0.f, 0.f, 0.f, 0.f, 0.f };
#pragma unroll
        for (int dc = 0; dc < 8; dc += 2) {
            const unsigned* bp0 = wsB + cb * 4096 + dc * 512 + lane * 16;
            const unsigned* bp1 = bp0 + 512;
            bf16x16 Bhi0 = *(const bf16x16*)bp0;
            bf16x16 Blo0 = *(const bf16x16*)(bp0 + 8);
            bf16x16 Bhi1 = *(const bf16x16*)bp1;
            bf16x16 Blo1 = *(const bf16x16*)(bp1 + 8);
            // chain 0: split-bf16 fp32 emulation hi*hi + hi*lo + lo*hi
            acc0 = __builtin_amdgcn_wmma_f32_16x16x32_bf16(
                       false, Ahi[dc], false, Bhi0, (short)0, acc0, false, false);
            // chain 1 interleaved
            acc1 = __builtin_amdgcn_wmma_f32_16x16x32_bf16(
                       false, Ahi[dc + 1], false, Bhi1, (short)0, acc1, false, false);
            acc0 = __builtin_amdgcn_wmma_f32_16x16x32_bf16(
                       false, Ahi[dc], false, Blo0, (short)0, acc0, false, false);
            acc1 = __builtin_amdgcn_wmma_f32_16x16x32_bf16(
                       false, Ahi[dc + 1], false, Blo1, (short)0, acc1, false, false);
            acc0 = __builtin_amdgcn_wmma_f32_16x16x32_bf16(
                       false, Alo[dc], false, Bhi0, (short)0, acc0, false, false);
            acc1 = __builtin_amdgcn_wmma_f32_16x16x32_bf16(
                       false, Alo[dc + 1], false, Bhi1, (short)0, acc1, false, false);
        }
        f32x8 acc = acc0 + acc1;

        const int   code = (cb << 4) + (lane & 15);
        const float bia  = wsBias[code];
#pragma unroll
        for (int jj = 0; jj < 8; ++jj) {
            float sc = acc[jj] - bia;        // z.e - 0.5||e||^2 (argmax)
            if (sc > bs[jj]) { bs[jj] = sc; bc[jj] = code; }
        }
    }

    // butterfly reduce over the 16 lanes sharing a row (halves stay disjoint)
#pragma unroll
    for (int m = 1; m < 16; m <<= 1) {
#pragma unroll
        for (int jj = 0; jj < 8; ++jj) {
            float rs = __shfl_xor(bs[jj], m, 32);
            int   rc = __shfl_xor(bc[jj], m, 32);
            if (rs > bs[jj] || (rs == bs[jj] && rc < bc[jj])) {
                bs[jj] = rs; bc[jj] = rc;
            }
        }
    }
    if ((lane & 15) == 0) {                  // lane 0 -> rows 0..7, lane 16 -> 8..15
        const int half = lane >> 4;
#pragma unroll
        for (int jj = 0; jj < 8; ++jj) {
            const int r    = strip * 16 + half * 8 + jj;
            const int code = bc[jj];
            sIdx[r] = code;
            outIdx[tileBase + r] = (float)code;
            atomicAdd(&counts[code], 1u);
        }
    }
    __syncthreads();

    // ---- Phase 3: gather z_q, straight-through output, loss
    float lsum = 0.f;
    {
        const int r     = t & 127;
        const int dHalf = t >> 7;
        const int code  = sIdx[r];
        const float* crow = cbw + (code << 8);
        const int zrow  = b << 8;
        for (int i = 0; i < 128; ++i) {
            const int d   = (i << 1) + dHalf;
            const int off = ((zrow + d) << 10) + hwBase + r;
            float zv   = z[off];
            float diff = crow[d] - zv;       // q - z
            outZq[off] = zv + diff;          // straight-through (== fp32 ref)
            lsum += diff * diff;
        }
    }
    sRed[t] = lsum;
    __syncthreads();
    for (int s = 128; s > 0; s >>= 1) {
        if (t < s) sRed[t] += sRed[t + s];
        __syncthreads();
    }
    if (t == 0) atomicAdd(lossAcc, sRed[0]);
}

// ---------------------------------------------------------------------------
// Kernel 3: perplexity + loss scalars
// ---------------------------------------------------------------------------
__global__ __launch_bounds__(256) void vq_final(
    const unsigned* __restrict__ counts, const float* __restrict__ lossAcc,
    float* __restrict__ outLoss, float* __restrict__ outPerp)
{
    __shared__ float red[256];
    float s = 0.f;
    for (int i = threadIdx.x; i < KCODES; i += 256) {
        float p = (float)counts[i] * (1.0f / (float)NROWS);
        s += p * __logf(p + 1e-10f);
    }
    red[threadIdx.x] = s;
    __syncthreads();
    for (int k = 128; k > 0; k >>= 1) {
        if (threadIdx.x < k) red[threadIdx.x] += red[threadIdx.x + k];
        __syncthreads();
    }
    if (threadIdx.x == 0) {
        *outPerp = __expf(-red[0]);
        *outLoss = BETA * (*lossAcc) * (1.0f / 16777216.0f); // mean over N*D
    }
}

extern "C" void kernel_launch(void* const* d_in, const int* in_sizes, int n_in,
                              void* d_out, int out_size, void* d_ws, size_t ws_size,
                              hipStream_t stream)
{
    const float* z   = (const float*)d_in[0];
    const float* cbw = (const float*)d_in[1];
    float* out     = (float*)d_out;
    float* outZq   = out;                       // 16777216
    float* outIdx  = out + 16777216;            // 65536
    float* outLoss = out + 16777216 + 65536;    // 1
    float* outPerp = outLoss + 1;               // 1

    unsigned* ws      = (unsigned*)d_ws;
    unsigned* wsB     = ws + WS_B;
    float*    wsBias  = (float*)(ws + WS_BIAS);
    unsigned* counts  = ws + WS_CNT;
    float*    lossAcc = (float*)(ws + WS_LOSS);

    vq_prep<<<64, 256, 0, stream>>>(cbw, wsB, wsBias, counts, lossAcc);
    vq_main<<<512, 256, (32768 + 128 + 256) * 4, stream>>>(
        z, cbw, wsB, wsBias, counts, lossAcc, outZq, outIdx);
    vq_final<<<1, 256, 0, stream>>>(counts, lossAcc, outLoss, outPerp);
}
